// conv_sknet_8022998909206
// MI455X (gfx1250) — matrix operand order repaired and checked by reading, not compiled, not run on hardware
//
#include <hip/hip_runtime.h>
#include <hip/hip_bf16.h>

#define NB   16
#define CIN  128
#define COUT 256
#define HH   64
#define WW   64
#define DG   32
#define YSZ  ((size_t)NB * COUT * HH * WW)   // 16,777,216 elements per branch

typedef __attribute__((ext_vector_type(16))) __bf16 v16bf;
typedef __attribute__((ext_vector_type(8)))  float  v8f;
typedef __attribute__((ext_vector_type(8)))  unsigned short u16x8;

union FragAB {
    u16x8 h[2];
    v16bf v;
};

static __device__ __forceinline__ unsigned short f32_to_bf16_rne(float f) {
    unsigned int u = __float_as_uint(f);
    unsigned int lsb = (u >> 16) & 1u;
    u += 0x7fffu + lsb;
    return (unsigned short)(u >> 16);
}

static __device__ __forceinline__ float bf16_bits_to_f32(unsigned short h) {
    return __uint_as_float(((unsigned int)h) << 16);
}

// ---------------------------------------------------------------------------
// Kernel A: implicit-GEMM conv (3x3, dilation dil, pad dil) via bf16x3 WMMA.
// Block = 256 threads (8 wave32 waves). Block tile: 64 co x 64 w (one full
// w-row at fixed (n,h)). K loop: 9 taps x 4 blocks of 32 input channels.
// Writes y = conv(x,w)+bias raw (pre-BN) and accumulates per-channel
// sum / sumsq for batch-stat BN.
// ---------------------------------------------------------------------------
__global__ __launch_bounds__(256) void conv_wmma_kernel(
    const float* __restrict__ x,      // [NB][CIN][HH][WW]
    const float* __restrict__ wgt,    // [COUT][CIN][3][3]
    const float* __restrict__ bias,   // [COUT]
    float* __restrict__ y,            // [NB][COUT][HH][WW]
    float* __restrict__ statSum,      // [COUT] (pre-zeroed)
    float* __restrict__ statSq,       // [COUT] (pre-zeroed)
    int dil)
{
    const int AP = 40;  // LDS pitch in halfwords: 80B rows -> 16B-aligned frags
    __shared__ __align__(16) unsigned short Ahi[64 * 40];
    __shared__ __align__(16) unsigned short Alo[64 * 40];
    __shared__ __align__(16) unsigned short Bhi[64 * 40];
    __shared__ __align__(16) unsigned short Blo[64 * 40];
    __shared__ float redS[64];
    __shared__ float redQ[64];

    const int tid  = threadIdx.x;
    const int bx   = blockIdx.x;       // n*HH + h
    const int n    = bx >> 6;
    const int h    = bx & 63;
    const int co0  = blockIdx.y * 64;

    if (tid < 64) { redS[tid] = 0.0f; redQ[tid] = 0.0f; }

    const int wave = tid >> 5;
    const int lane = tid & 31;
    const int mt   = wave & 3;   // M subtile (16 co)
    const int nh   = wave >> 2;  // N subtile pair (2 x 16 w)

    v8f acc0 = {};
    v8f acc1 = {};

    for (int tap = 0; tap < 9; ++tap) {
        const int dh = dil * ((tap / 3) - 1);
        const int dw = dil * ((tap % 3) - 1);
        const int hs = h + dh;
        const bool hok = (hs >= 0) && (hs < HH);

        for (int cb = 0; cb < 4; ++cb) {
            const int ci0 = cb * 32;
            __syncthreads();  // protect LDS tiles from previous iteration

            // ---- stage B tile: x[n][ci0..+31][hs][w+dw] -> Bt[w][ci] hi/lo
            {
                const int ci = tid >> 3;            // 0..31
                const int w0 = (tid & 7) * 8;       // 8 consecutive w each
                const float* src =
                    x + (((size_t)n * CIN + ci0 + ci) * HH + (hok ? hs : 0)) * WW;
#pragma unroll
                for (int j = 0; j < 8; ++j) {
                    const int w  = w0 + j;
                    const int wsrc = w + dw;
                    const float v =
                        (hok && wsrc >= 0 && wsrc < WW) ? src[wsrc] : 0.0f;
                    const unsigned short hb = f32_to_bf16_rne(v);
                    const unsigned short lb =
                        f32_to_bf16_rne(v - bf16_bits_to_f32(hb));
                    Bhi[w * AP + ci] = hb;
                    Blo[w * AP + ci] = lb;
                }
            }
            // ---- stage A tile: wgt[co0..+63][ci0..+31][tap] -> A[co][ci] hi/lo
            {
                const int co_l  = tid >> 2;          // 0..63
                const int cbase = (tid & 3) * 8;     // 8 ci each
                const float* src = wgt + (size_t)(co0 + co_l) * CIN * 9 + tap;
#pragma unroll
                for (int j = 0; j < 8; ++j) {
                    const int ci = cbase + j;
                    const float v = src[(size_t)(ci0 + ci) * 9];
                    const unsigned short hb = f32_to_bf16_rne(v);
                    const unsigned short lb =
                        f32_to_bf16_rne(v - bf16_bits_to_f32(hb));
                    Ahi[co_l * AP + ci] = hb;
                    Alo[co_l * AP + ci] = lb;
                }
            }
            __syncthreads();

            // ---- A fragment (16x32 bf16, documented lane layout):
            // lanes 0-15: halves 0-7 = K 0-7, halves 8-15 = K 16-23
            // lanes 16-31: halves 0-7 = K 8-15, halves 8-15 = K 24-31
            const int m    = mt * 16 + (lane & 15);
            const int koff = (lane >= 16) ? 8 : 0;
            FragAB ah, al;
            ah.h[0] = *(const u16x8*)&Ahi[m * AP + koff];
            ah.h[1] = *(const u16x8*)&Ahi[m * AP + 16 + koff];
            al.h[0] = *(const u16x8*)&Alo[m * AP + koff];
            al.h[1] = *(const u16x8*)&Alo[m * AP + 16 + koff];

            // ---- B fragments (32x16 bf16): lane L -> N = L%16,
            // halves 0-7 = K koff2+0..7, halves 8-15 = K 16+koff2+0..7 (koff2 = L<16 ? 0 : 8)
            const int koff2 = (lane >= 16) ? 8 : 0;
#pragma unroll
            for (int s = 0; s < 2; ++s) {
                const int wl = (nh * 2 + s) * 16 + (lane & 15);
                FragAB bh, bl;
                bh.h[0] = *(const u16x8*)&Bhi[wl * AP + koff2];
                bh.h[1] = *(const u16x8*)&Bhi[wl * AP + koff2 + 16];
                bl.h[0] = *(const u16x8*)&Blo[wl * AP + koff2];
                bl.h[1] = *(const u16x8*)&Blo[wl * AP + koff2 + 16];

                v8f acc = s ? acc1 : acc0;
                // bf16x3 split product: hi*hi + hi*lo + lo*hi (f32 accum)
                acc = __builtin_amdgcn_wmma_f32_16x16x32_bf16(
                        false, ah.v, false, bh.v, (short)0, acc, false, false);
                acc = __builtin_amdgcn_wmma_f32_16x16x32_bf16(
                        false, ah.v, false, bl.v, (short)0, acc, false, false);
                acc = __builtin_amdgcn_wmma_f32_16x16x32_bf16(
                        false, al.v, false, bh.v, (short)0, acc, false, false);
                if (s) acc1 = acc; else acc0 = acc;
            }
        }
    }

    // ---- epilogue: add bias, store raw y, accumulate BN stats
#pragma unroll
    for (int s = 0; s < 2; ++s) {
        const v8f acc = s ? acc1 : acc0;
        const int wl = (nh * 2 + s) * 16 + (lane & 15);
#pragma unroll
        for (int r = 0; r < 8; ++r) {
            // C/D layout: VGPR r -> M = r (lanes 0-15) / r+8 (lanes 16-31)
            const int ml = mt * 16 + r + ((lane >= 16) ? 8 : 0);
            const int co = co0 + ml;
            const float v = acc[r] + bias[co];
            y[(((size_t)n * COUT + co) * HH + h) * WW + wl] = v;
            atomicAdd(&redS[ml], v);
            atomicAdd(&redQ[ml], v * v);
        }
    }
    __syncthreads();
    if (tid < 64) {
        atomicAdd(&statSum[co0 + tid], redS[tid]);
        atomicAdd(&statSq[co0 + tid], redQ[tid]);
    }
}

// ---------------------------------------------------------------------------
__global__ void zero_kernel(float* p, int n) {
    int i = blockIdx.x * blockDim.x + threadIdx.x;
    if (i < n) p[i] = 0.0f;
}

__global__ void bn_finalize_kernel(
    const float* __restrict__ sum3, const float* __restrict__ sq3,
    const float* __restrict__ sum5, const float* __restrict__ sq5,
    const float* __restrict__ g3, const float* __restrict__ be3,
    const float* __restrict__ g5, const float* __restrict__ be5,
    float* __restrict__ sc3, float* __restrict__ sh3,
    float* __restrict__ sc5, float* __restrict__ sh5)
{
    int c = blockIdx.x * blockDim.x + threadIdx.x;
    if (c >= COUT) return;
    const float inv = 1.0f / (float)(NB * HH * WW);
    {
        float mu  = sum3[c] * inv;
        float var = fmaxf(sq3[c] * inv - mu * mu, 0.0f);
        float sc  = g3[c] * rsqrtf(var + 1e-5f);
        sc3[c] = sc;
        sh3[c] = be3[c] - mu * sc;
    }
    {
        float mu  = sum5[c] * inv;
        float var = fmaxf(sq5[c] * inv - mu * mu, 0.0f);
        float sc  = g5[c] * rsqrtf(var + 1e-5f);
        sc5[c] = sc;
        sh5[c] = be5[c] - mu * sc;
    }
}

// s[b][c] = mean_{h,w}( relu(bn3(y3)) + relu(bn5(y5)) )
__global__ __launch_bounds__(256) void pool_kernel(
    const float* __restrict__ y3, const float* __restrict__ y5,
    const float* __restrict__ sc3, const float* __restrict__ sh3,
    const float* __restrict__ sc5, const float* __restrict__ sh5,
    float* __restrict__ s)
{
    const int bc = blockIdx.x;        // n*COUT + c
    const int c  = bc & (COUT - 1);
    const float a3 = sc3[c], o3 = sh3[c];
    const float a5 = sc5[c], o5 = sh5[c];
    const float* p3 = y3 + (size_t)bc * (HH * WW);
    const float* p5 = y5 + (size_t)bc * (HH * WW);

    float acc = 0.0f;
    for (int i = threadIdx.x; i < HH * WW; i += 256)
        acc += fmaxf(p3[i] * a3 + o3, 0.0f) + fmaxf(p5[i] * a5 + o5, 0.0f);

    __shared__ float red[256];
    red[threadIdx.x] = acc;
    __syncthreads();
    for (int off = 128; off > 0; off >>= 1) {
        if (threadIdx.x < off) red[threadIdx.x] += red[threadIdx.x + off];
        __syncthreads();
    }
    if (threadIdx.x == 0) s[bc] = red[0] * (1.0f / (float)(HH * WW));
}

// z = relu(s @ fc_w^T + fc_b);  gate = sigmoid(z @ (A - B))
__global__ __launch_bounds__(512) void gate_kernel(
    const float* __restrict__ s, const float* __restrict__ fcw,
    const float* __restrict__ fcb, const float* __restrict__ Amat,
    const float* __restrict__ Bmat, float* __restrict__ gate)
{
    __shared__ float z[NB * DG];  // 512
    const int t = threadIdx.x;
    {
        const int b = t >> 5;     // 0..15
        const int d = t & 31;     // 0..31
        float acc = fcb[d];
        const float* sp = s + (size_t)b * COUT;
        const float* wp = fcw + (size_t)d * COUT;
        for (int c = 0; c < COUT; ++c) acc += sp[c] * wp[c];
        z[b * DG + d] = fmaxf(acc, 0.0f);
    }
    __syncthreads();
    for (int i = t; i < NB * COUT; i += 512) {
        const int b = i >> 8;
        const int c = i & (COUT - 1);
        float acc = 0.0f;
        const float* zp = z + b * DG;
        for (int d = 0; d < DG; ++d)
            acc += zp[d] * (Amat[(size_t)d * COUT + c] - Bmat[(size_t)d * COUT + c]);
        gate[i] = 1.0f / (1.0f + __expf(-acc));
    }
}

// out = relu(bn3(y3))*g + relu(bn5(y5))*(1-g), float4-vectorized along w
__global__ __launch_bounds__(256) void blend_kernel(
    const float* __restrict__ y3, const float* __restrict__ y5,
    const float* __restrict__ sc3, const float* __restrict__ sh3,
    const float* __restrict__ sc5, const float* __restrict__ sh5,
    const float* __restrict__ gate, float* __restrict__ out)
{
    const size_t i = (size_t)blockIdx.x * 256 + threadIdx.x;
    const size_t total4 = YSZ / 4;
    if (i >= total4) return;
    const int bc = (int)((i * 4) >> 12);   // /(HH*WW)
    const int c  = bc & (COUT - 1);
    const float a3 = sc3[c], o3 = sh3[c];
    const float a5 = sc5[c], o5 = sh5[c];
    const float g  = gate[bc];
    const float4 v3 = ((const float4*)y3)[i];
    const float4 v5 = ((const float4*)y5)[i];
    float4 o;
    o.x = fmaxf(v3.x * a3 + o3, 0.0f) * g + fmaxf(v5.x * a5 + o5, 0.0f) * (1.0f - g);
    o.y = fmaxf(v3.y * a3 + o3, 0.0f) * g + fmaxf(v5.y * a5 + o5, 0.0f) * (1.0f - g);
    o.z = fmaxf(v3.z * a3 + o3, 0.0f) * g + fmaxf(v5.z * a5 + o5, 0.0f) * (1.0f - g);
    o.w = fmaxf(v3.w * a3 + o3, 0.0f) * g + fmaxf(v5.w * a5 + o5, 0.0f) * (1.0f - g);
    ((float4*)out)[i] = o;
}

// ---------------------------------------------------------------------------
extern "C" void kernel_launch(void* const* d_in, const int* in_sizes, int n_in,
                              void* d_out, int out_size, void* d_ws, size_t ws_size,
                              hipStream_t stream) {
    (void)in_sizes; (void)n_in; (void)out_size; (void)ws_size;
    const float* x   = (const float*)d_in[0];
    const float* w3  = (const float*)d_in[1];
    const float* b3  = (const float*)d_in[2];
    const float* g3  = (const float*)d_in[3];
    const float* be3 = (const float*)d_in[4];
    const float* w5  = (const float*)d_in[5];
    const float* b5  = (const float*)d_in[6];
    const float* g5  = (const float*)d_in[7];
    const float* be5 = (const float*)d_in[8];
    const float* fcw = (const float*)d_in[9];
    const float* fcb = (const float*)d_in[10];
    const float* Am  = (const float*)d_in[11];
    const float* Bm  = (const float*)d_in[12];
    float* out = (float*)d_out;

    // workspace layout (floats)
    float* ws   = (float*)d_ws;
    float* y3   = ws;                 // YSZ
    float* y5   = y3 + YSZ;           // YSZ
    float* sum3 = y5 + YSZ;           // 256
    float* sq3  = sum3 + COUT;
    float* sum5 = sq3 + COUT;
    float* sq5  = sum5 + COUT;
    float* sc3  = sq5 + COUT;
    float* sh3  = sc3 + COUT;
    float* sc5  = sh3 + COUT;
    float* sh5  = sc5 + COUT;
    float* sbuf = sh5 + COUT;         // NB*COUT
    float* gbuf = sbuf + NB * COUT;   // NB*COUT

    // zero BN stat accumulators (graph-capture-safe: plain kernel)
    zero_kernel<<<(4 * COUT + 255) / 256, 256, 0, stream>>>(sum3, 4 * COUT);

    dim3 gA(NB * HH, COUT / 64, 1);   // 1024 x 4
    conv_wmma_kernel<<<gA, 256, 0, stream>>>(x, w3, b3, y3, sum3, sq3, 1);
    conv_wmma_kernel<<<gA, 256, 0, stream>>>(x, w5, b5, y5, sum5, sq5, 2);

    bn_finalize_kernel<<<1, COUT, 0, stream>>>(sum3, sq3, sum5, sq5,
                                               g3, be3, g5, be5,
                                               sc3, sh3, sc5, sh5);

    pool_kernel<<<NB * COUT, 256, 0, stream>>>(y3, y5, sc3, sh3, sc5, sh5, sbuf);

    gate_kernel<<<1, 512, 0, stream>>>(sbuf, fcw, fcb, Am, Bm, gbuf);

    blend_kernel<<<(unsigned)((YSZ / 4 + 255) / 256), 256, 0, stream>>>(
        y3, y5, sc3, sh3, sc5, sh5, gbuf, out);
}
